// final_conv_38912403702000
// MI455X (gfx1250) — matrix operand order repaired and checked by reading, not compiled, not run on hardware
//
#include <hip/hip_runtime.h>
#include <hip/hip_bf16.h>
#include <stdint.h>

// ---------------------------------------------------------------------------
// 4D conv (N=1,Cin=16,Cout=4, T=10,D=32,H=64,W=64, k=3^4, pad=1) + channel
// softmax as implicit GEMM on the CDNA5 WMMA pipe, shifted-window version:
//   A (16x32 bf16): M rows = (dx in 0..3, co in 0..3);
//                   A[(dx,co)][cin,kt,kz,ky,wx] = W[co][cin,kt,kz,ky,wx-dx]
//                   (zero when wx-dx outside 0..2)
//   B (32x16 bf16): columns = 16 base x positions (stride 4);
//                   B[k][n] = xpad[cin, t+kt, z+kz, y+ky, 4n+wx]
//   D (16x16 f32) : row (dx,co), col n -> logit for x = 4n+dx, channel co.
// One wave = one full 64-wide x row x 4 Cout (256 outputs) in 81 WMMAs.
// K = 16*27*6 = 2592 (81 chunks of 32, no padding); wx period 6 is even so
// every K-pair is x-contiguous -> one aligned b32 load of a pre-packed
// bf16 pair per B VGPR.
// ---------------------------------------------------------------------------

typedef __attribute__((ext_vector_type(16))) __bf16 v16bf;
typedef __attribute__((ext_vector_type(8)))  float  v8f;

// padded input geometry: [cin][tp=12][zp=34][yp=66][xp=72]
#define XP     72
#define S_Y    72
#define S_Z    (66 * 72)              // 4752
#define S_T    (34 * S_Z)             // 161568
#define S_C    (12 * S_T)             // 1938816
#define NPAD   (16 * S_C)             // 31021056 u32-packed bf16 pairs

#define KTOT   2592                   // 16*3*3*3*6
#define NCHUNK 81                     // 2592 / 32
#define CS_OUT (10 * 32 * 64 * 64)    // per-channel output plane
#define LDS_BYTES ((NCHUNK * 256 + NCHUNK * 16) * 4)   // 88128

union Frag {
    int4     q[2];
    uint32_t u[8];
    v16bf    v;
};

__device__ __forceinline__ uint32_t f2bf(float f) {
    uint32_t u = __float_as_uint(f);
    return (u + 0x7FFFu + ((u >> 16) & 1u)) >> 16;   // RNE
}

// -------- prep 1: padded, pair-packed bf16 input ---------------------------
__global__ void pack_x_kernel(const float* __restrict__ x,
                              uint32_t* __restrict__ xp) {
    uint32_t e = blockIdx.x * 256u + threadIdx.x;
    if (e >= (uint32_t)NPAD) return;
    uint32_t xc = e % XP;  uint32_t r = e / XP;
    uint32_t yc = r % 66;  r /= 66;
    uint32_t zc = r % 34;  r /= 34;
    uint32_t tc = r % 12;  uint32_t ci = r / 12;
    float v0 = 0.f, v1 = 0.f;
    if (yc >= 1 && yc <= 64 && zc >= 1 && zc <= 32 && tc >= 1 && tc <= 10) {
        const float* row = x + ((((ci * 10u + (tc - 1u)) * 32u + (zc - 1u)) * 64u
                               + (yc - 1u)) * 64u);
        if (xc >= 1 && xc <= 64) v0 = row[xc - 1u];
        if (xc + 1 <= 64) v1 = row[xc];                  // (xc+1)-1 = xc
    }
    xp[e] = f2bf(v0) | (f2bf(v1) << 16);
}

// k = (((cin*3+kt)*3+kz)*3+ky)*6 + wx
__device__ __forceinline__ void kdecomp(int kk, int& cin, int& kt, int& kz,
                                        int& ky, int& wx) {
    wx = kk % 6;
    int r = kk / 6;
    ky = r % 3; r /= 3;
    kz = r % 3; r /= 3;
    kt = r % 3;
    cin = r / 3;
}

__device__ __forceinline__ float wval(const float* __restrict__ W, int m, int kk) {
    int cin, kt, kz, ky, wx;
    kdecomp(kk, cin, kt, kz, ky, wx);
    int dx = m >> 2, co = m & 3;
    int kx = wx - dx;
    if (kx < 0 || kx > 2) return 0.f;
    return W[((((co * 16 + cin) * 3 + kt) * 3 + kz) * 3 + ky) * 3 + kx];
}

// -------- prep 2: A fragments (exact WMMA lane layout) + B offset table ----
// A 16-bit 16x32 layout: lanes 0-15 (M=lane): v0..3 -> K pairs {0..7},
// v4..7 -> {16..23}; lanes 16-31 (M=lane-16): {8..15}, {24..31}.
// B 16-bit 32x16 layout: lanes 0-15 (col=lane): v0..3 -> {0..7}, v4..7 -> {16..23}; lanes 16-31:
// {8..15}, {24..31} (same K order as A).
__global__ void pack_wA_kernel(const float* __restrict__ W,
                               uint32_t* __restrict__ Apack,
                               int* __restrict__ offs) {
    int c = blockIdx.x;                 // chunk 0..80
    int t = threadIdx.x;                // 0..255
    int l = t >> 3, v = t & 7;
    int kk = 32 * c + ((v < 4) ? 2 * v : 8 + 2 * v) + ((l < 16) ? 0 : 8);
    int m  = l & 15;
    uint32_t lo = f2bf(wval(W, m, kk));
    uint32_t hi = f2bf(wval(W, m, kk + 1));
    Apack[(c * 32 + l) * 8 + v] = lo | (hi << 16);

    if (t < 16) {                       // offset table for B gathers
        int half = t >> 3, vv = t & 7;
        int kb = 32 * c + ((vv < 4) ? 2 * vv : 8 + 2 * vv) + 8 * half;
        int cin, kt, kz, ky, wx;
        kdecomp(kb, cin, kt, kz, ky, wx);
        offs[c * 16 + t] = cin * S_C + kt * S_T + kz * S_Z + ky * S_Y + wx;
    }
}

// -------- main: 1 wave = one 64-wide x row x 4 Cout, 81 bf16 WMMAs ---------
__global__ __launch_bounds__(256) void conv4d_wmma_kernel(
        const uint32_t* __restrict__ xp,
        const uint32_t* __restrict__ Apack,
        const int* __restrict__ offsg,
        const float* __restrict__ bias,
        float* __restrict__ out) {
    extern __shared__ uint32_t smem[];
    uint32_t* sA = smem;                       // NCHUNK*256 u32 (82944 B)
    int*      sO = (int*)(smem + NCHUNK * 256);// NCHUNK*16  i32 ( 5184 B)

    // cooperative stage of A fragments + offset table (b128 copies)
    for (int i = threadIdx.x; i < NCHUNK * 64; i += 256)
        ((int4*)sA)[i] = ((const int4*)Apack)[i];
    for (int i = threadIdx.x; i < NCHUNK * 4; i += 256)
        ((int4*)sO)[i] = ((const int4*)offsg)[i];
    __syncthreads();

    const int lane = threadIdx.x & 31;
    const int wv   = threadIdx.x >> 5;
    const int tile = blockIdx.x * 8 + wv;          // 20480 tiles = (t,z,y) rows
    const int y  = tile & 63;
    const int z  = (tile >> 6) & 31;
    const int tt = tile >> 11;                     // 0..9
    const int col  = lane & 15;                    // base position n (x = 4n)
    const int half = lane >> 4;

    const uint32_t* bp = xp + (tt * S_T + z * S_Z + y * S_Y + 4 * col);

    v8f acc = {0.f, 0.f, 0.f, 0.f, 0.f, 0.f, 0.f, 0.f};

#pragma unroll 3
    for (int c = 0; c < NCHUNK; ++c) {
        Frag a, b;
        const int4* ap = (const int4*)(sA + (c * 32 + lane) * 8);
        a.q[0] = ap[0];
        a.q[1] = ap[1];
        const int4* op = (const int4*)(sO + c * 16 + half * 8);
        int4 o0 = op[0];
        int4 o1 = op[1];
        b.u[0] = bp[o0.x]; b.u[1] = bp[o0.y];
        b.u[2] = bp[o0.z]; b.u[3] = bp[o0.w];
        b.u[4] = bp[o1.x]; b.u[5] = bp[o1.y];
        b.u[6] = bp[o1.z]; b.u[7] = bp[o1.w];
        acc = __builtin_amdgcn_wmma_f32_16x16x32_bf16(
                  false, a.v, false, b.v, (short)0, acc, false, false);
    }

    // D layout: lane (half,col) holds rows M = 8*half + r in v_r, col N = col.
    // M = dx*4 + co  ->  v0..3 = (dx=2*half+0, co=0..3),
    //                    v4..7 = (dx=2*half+1, co=0..3).  x_out = 4*col + dx.
    float* orow = out + (((tt * 32 + z) * 64 + y) << 6) + 4 * col;
    const float b0 = bias[0], b1 = bias[1], b2 = bias[2], b3 = bias[3];
#pragma unroll
    for (int j = 0; j < 2; ++j) {
        const int dx = 2 * half + j;
        float l0 = acc[4 * j + 0] + b0;
        float l1 = acc[4 * j + 1] + b1;
        float l2 = acc[4 * j + 2] + b2;
        float l3 = acc[4 * j + 3] + b3;
        float mx = fmaxf(fmaxf(l0, l1), fmaxf(l2, l3));
        float e0 = __expf(l0 - mx), e1 = __expf(l1 - mx);
        float e2 = __expf(l2 - mx), e3 = __expf(l3 - mx);
        float inv = 1.f / (e0 + e1 + e2 + e3);
        float* o = orow + dx;
        o[0 * CS_OUT] = e0 * inv;
        o[1 * CS_OUT] = e1 * inv;
        o[2 * CS_OUT] = e2 * inv;
        o[3 * CS_OUT] = e3 * inv;
    }
}

extern "C" void kernel_launch(void* const* d_in, const int* in_sizes, int n_in,
                              void* d_out, int out_size, void* d_ws, size_t ws_size,
                              hipStream_t stream) {
    const float* x = (const float*)d_in[0];
    const float* W = (const float*)d_in[1];
    const float* b = (const float*)d_in[2];
    float* out = (float*)d_out;

    uint32_t* xp    = (uint32_t*)d_ws;               // NPAD u32
    uint32_t* Apack = xp + NPAD;                     // NCHUNK*256 u32
    int*      offs  = (int*)(Apack + NCHUNK * 256);  // NCHUNK*16 i32

    pack_x_kernel<<<(NPAD + 255) / 256, 256, 0, stream>>>(x, xp);
    pack_wA_kernel<<<NCHUNK, 256, 0, stream>>>(W, Apack, offs);
    conv4d_wmma_kernel<<<2560, 256, LDS_BYTES, stream>>>(xp, Apack, offs, b, out);
}
